// MultiHeadAttention_89524298317897
// MI455X (gfx1250) — matrix operand order repaired and checked by reading, not compiled, not run on hardware
//
#include <hip/hip_runtime.h>
#include <hip/hip_bf16.h>
#include <math.h>

// ---------------------------------------------------------------------------
// MHA forward for MI455X (gfx1250), bf16 WMMA + CDNA5 async/TDM staging.
//   B=8, S=1024, D_MODEL=1024, NHEAD=16, D_HEAD=64
// Stages:
//   1) convert fp32 inputs/weights -> bf16 scratch
//   2) Qp/Kp/Vp = x @ W^T : WMMA GEMM, weight tile double-buffered in LDS
//      via global_load_async_to_lds_b128 (ASYNCcnt)
//   3) flash attention: K/V tiles staged to LDS via TDM tensor_load_to_lds
//      (TENSORcnt), scores + PV via v_wmma_f32_16x16x32_bf16
//   4) out = ctx @ w_o^T (fp32 output)
// ---------------------------------------------------------------------------

typedef __attribute__((ext_vector_type(16))) __bf16    bf16x16;
typedef __attribute__((ext_vector_type(8)))  __bf16    bf16x8;
typedef __attribute__((ext_vector_type(8)))  float     f32x8;
typedef __attribute__((ext_vector_type(4)))  unsigned  ui32x4;
typedef __attribute__((ext_vector_type(8)))  unsigned  ui32x8;

#define B_SZ     8
#define S_LEN    1024
#define DMODEL   1024
#define NHEAD    16
#define DHEAD    64
#define MROWS    (B_SZ * S_LEN)          // 8192
#define HEADELEM (S_LEN * DHEAD)         // 65536 elems per (b,h) chunk

// ---- WMMA helper -----------------------------------------------------------
__device__ __forceinline__ f32x8 wmma_bf16(bf16x16 a, bf16x16 b, f32x8 c) {
  return __builtin_amdgcn_wmma_f32_16x16x32_bf16(false, a, false, b,
                                                 (short)0, c, false, false);
}

// A-fragment: 16x32 bf16 tile, row-major, leading dim ld.
// lanes 0-15: M=lane, K=0..15 contiguous; lanes 16-31: M=lane-16, K=16..31.
__device__ __forceinline__ bf16x16 load_a_frag(const __bf16* base, int ld) {
  int lane = threadIdx.x & 31;
  int row  = lane & 15;
  int kh   = (lane >> 4) << 4;
  const __bf16* p = base + (size_t)row * ld + kh;
  bf16x8 lo = *(const bf16x8*)(p);
  bf16x8 hi = *(const bf16x8*)(p + 8);
  bf16x16 r;
#pragma unroll
  for (int i = 0; i < 8; ++i) { r[i] = lo[i]; r[i + 8] = hi[i]; }
  return r;
}

// B-fragment (32x16) from a transposed-B matrix stored N x K row-major.
// lanes 0-15: N=lane, K=0..15 contiguous; lanes 16-31: N=lane-16, K=16..31.
__device__ __forceinline__ bf16x16 load_bt_frag(const __bf16* base, int ld) {
  int lane = threadIdx.x & 31;
  int col  = lane & 15;
  int kh   = (lane >> 4) << 4;
  return *(const bf16x16*)(base + (size_t)col * ld + kh);
}

// ---- CDNA5 async copy: one b128 per lane, global -> LDS (ASYNCcnt) ---------
__device__ __forceinline__ void async_copy_b128(unsigned lds_addr,
                                                const void* gaddr) {
  asm volatile("global_load_async_to_lds_b128 %0, %1, off"
               :: "v"(lds_addr), "v"((unsigned long long)(size_t)gaddr)
               : "memory");
}
__device__ __forceinline__ void wait_asynccnt0() {
  asm volatile("s_wait_asynccnt 0x0" ::: "memory");
}

// ---- CDNA5 TDM: 2-D tile load, global -> LDS (TENSORcnt) -------------------
// D# groups per CDNA5 ISA. 2-D tensor -> only
// groups 0 and 1 (VADDR2/3 omitted). tensor dims == tile dims (no OOB).
__device__ __forceinline__ void tdm_load_2d(unsigned lds_addr, const void* gaddr,
                                            unsigned d0,      // elems per row
                                            unsigned d1,      // rows
                                            unsigned stride0) // row stride, elems
{
  unsigned long long ga = (unsigned long long)(size_t)gaddr;
  ui32x4 g0;
  g0[0] = 1u;                                          // count=1, user D#
  g0[1] = lds_addr;                                    // lds_addr [63:32]
  g0[2] = (unsigned)(ga & 0xffffffffu);                // global_addr [95:64]
  g0[3] = (unsigned)((ga >> 32) & 0x01ffffffu)         // global_addr [120:96]
        | (2u << 30);                                  // type=2 ("image")
  ui32x8 g1;
  g1[0] = (1u << 16);                                  // data_size=1 (2 bytes)
  g1[1] = (d0 & 0xffffu) << 16;                        // tensor_dim0[15:0]
  g1[2] = ((d0 >> 16) & 0xffffu)                       // tensor_dim0[31:16]
        | ((d1 & 0xffffu) << 16);                      // tensor_dim1[15:0]
  g1[3] = ((d1 >> 16) & 0xffffu)                       // tensor_dim1[31:16]
        | ((d0 & 0xffffu) << 16);                      // tile_dim0
  g1[4] = (d1 & 0xffffu);                              // tile_dim1 (tile_dim2=0)
  g1[5] = stride0;                                     // tensor_dim0_stride lo32
  g1[6] = 0u;                                          // stride0 hi / stride1 lo
  g1[7] = 0u;
  asm volatile("tensor_load_to_lds %0, %1" :: "s"(g0), "s"(g1) : "memory");
}
__device__ __forceinline__ void wait_tensorcnt0() {
  asm volatile("s_wait_tensorcnt 0x0" ::: "memory");
}

// ---- Stage 1: fp32 -> bf16 -------------------------------------------------
__global__ void cvt_f32_to_bf16(const float* __restrict__ src,
                                __bf16* __restrict__ dst, int n) {
  int i = blockIdx.x * blockDim.x + threadIdx.x;
  if (i < n) dst[i] = (__bf16)src[i];
}

// ---- Stage 2/4: C[M,N] = A[M,K] @ W[N,K]^T ---------------------------------
// 256 threads = 8 waves; wave w: 16(M) x 64(N).  Block tile: 128M x 64N.
// W tile (64N x 32K, 4KB) double-buffered in LDS via async-to-LDS; 8x reuse.
__global__ void gemm_xwt(const __bf16* __restrict__ A,
                         const __bf16* __restrict__ W,
                         float* __restrict__ Cf, __bf16* __restrict__ Cb,
                         int M, int N, int K, int store_bf16) {
  __shared__ __align__(32) __bf16 wtile[2][64 * 32];

  int wv   = threadIdx.x >> 5;
  int lane = threadIdx.x & 31;
  int m0 = blockIdx.x * 128 + wv * 16;
  int n0 = blockIdx.y * 64;

  int t    = threadIdx.x;
  int lrow = t >> 2;            // 0..63 : N row of weight tile
  int lseg = (t & 3) * 8;       // 0,8,16,24 : K segment (8 bf16 = 16B)

  auto issue_wtile = [&](int k0, int buf) {
    const __bf16* g = W + (size_t)(n0 + lrow) * K + k0 + lseg;
    unsigned lds = (unsigned)(size_t)&wtile[buf][lrow * 32 + lseg];
    async_copy_b128(lds, g);
  };

  issue_wtile(0, 0);

  f32x8 acc[4] = {};
  int buf = 0;
  for (int k0 = 0; k0 < K; k0 += 32, buf ^= 1) {
    wait_asynccnt0();
    __syncthreads();                      // W tile `buf` resident for all waves
    if (k0 + 32 < K) {
      issue_wtile(k0 + 32, buf ^ 1);      // overlap next stage with compute
      __builtin_prefetch(A + (size_t)m0 * K + k0 + 32, 0, 1);
    }
    bf16x16 a = load_a_frag(A + (size_t)m0 * K + k0, K);
    const __bf16* wt = wtile[buf];
#pragma unroll
    for (int s = 0; s < 4; ++s) {
      bf16x16 b = load_bt_frag(wt + (size_t)(s * 16) * 32, 32);
      acc[s] = wmma_bf16(a, b, acc[s]);
    }
    __syncthreads();                      // all waves done before overwrite
  }

  int col  = lane & 15;
  int half = lane >> 4;
#pragma unroll
  for (int s = 0; s < 4; ++s) {
#pragma unroll
    for (int r = 0; r < 8; ++r) {
      size_t idx = (size_t)(m0 + r + 8 * half) * N + (n0 + s * 16 + col);
      if (store_bf16) Cb[idx] = (__bf16)acc[s][r];
      else            Cf[idx] = acc[s][r];
    }
  }
}

// ---- Stage 3: flash attention ---------------------------------------------
// Block = 4 waves = one (b,h) chunk x 64 query rows; wave owns 16 rows.
// Per 32-key block: TDM stages K(32x64) and V(32x64) tiles into LDS (shared by
// all 4 waves), 4 score WMMAs, online softmax, LDS P re-layout, 4 PV WMMAs.
__global__ void flash_attn(const __bf16* __restrict__ Qp,
                           const __bf16* __restrict__ Kp,
                           const __bf16* __restrict__ Vp,
                           const unsigned char* __restrict__ mask,
                           __bf16* __restrict__ ctx) {
  __shared__ __align__(32) __bf16 ktile[32 * 64];       // 4KB
  __shared__ __align__(32) __bf16 vtile[32 * 64];       // 4KB
  __shared__ __align__(32) __bf16 pshare[4][16 * 32];   // 4KB, per-wave

  int wv   = threadIdx.x >> 5;
  int lane = threadIdx.x & 31;
  int bh   = blockIdx.x >> 4;          // b*16 + h  (chunked view)
  int qt   = blockIdx.x & 15;
  int b    = bh >> 4;

  const __bf16* Q  = Qp + (size_t)bh * HEADELEM;
  const __bf16* Kb = Kp + (size_t)bh * HEADELEM;
  const __bf16* Vb = Vp + (size_t)bh * HEADELEM;
  const unsigned char* mk = mask + (size_t)b * S_LEN;

  int q0 = qt * 64 + wv * 16;
  bf16x16 qa0 = load_a_frag(Q + (size_t)q0 * DHEAD +  0, DHEAD);
  bf16x16 qa1 = load_a_frag(Q + (size_t)q0 * DHEAD + 32, DHEAD);

  f32x8 o[4] = {};
  float m_r[8], l_r[8];
#pragma unroll
  for (int r = 0; r < 8; ++r) { m_r[r] = -INFINITY; l_r[r] = 0.0f; }

  const float scale = 0.125f;          // 1/sqrt(64)
  int   col  = lane & 15;
  __bf16* pl = pshare[wv];
  unsigned kt_lds = (unsigned)(size_t)&ktile[0];
  unsigned vt_lds = (unsigned)(size_t)&vtile[0];

  for (int k0 = 0; k0 < S_LEN; k0 += 32) {
    // ---- TDM: stage K/V 32x64 bf16 tiles into LDS (wave 0 drives DMA) ----
    if (wv == 0) {
      tdm_load_2d(kt_lds, Kb + (size_t)k0 * DHEAD, DHEAD, 32, DHEAD);
      tdm_load_2d(vt_lds, Vb + (size_t)k0 * DHEAD, DHEAD, 32, DHEAD);
      wait_tensorcnt0();
    }
    __syncthreads();

    // ---- scores: two 16x16 key blocks, K-dim = DHEAD split in 2 ----
    f32x8 s0 = {}, s1 = {};
    {
      bf16x16 b00 = load_bt_frag(ktile +  0 * DHEAD +  0, DHEAD);
      bf16x16 b01 = load_bt_frag(ktile +  0 * DHEAD + 32, DHEAD);
      s0 = wmma_bf16(qa0, b00, s0);
      s0 = wmma_bf16(qa1, b01, s0);
      bf16x16 b10 = load_bt_frag(ktile + 16 * DHEAD +  0, DHEAD);
      bf16x16 b11 = load_bt_frag(ktile + 16 * DHEAD + 32, DHEAD);
      s1 = wmma_bf16(qa0, b10, s1);
      s1 = wmma_bf16(qa1, b11, s1);
    }

    // ---- scale + key-padding mask (key index = chunk index per reference)
    bool msk0 = mk[k0 + col]      != 0;
    bool msk1 = mk[k0 + 16 + col] != 0;
    float rowm[8];
#pragma unroll
    for (int r = 0; r < 8; ++r) {
      float v0 = msk0 ? -INFINITY : s0[r] * scale;
      float v1 = msk1 ? -INFINITY : s1[r] * scale;
      s0[r] = v0; s1[r] = v1;
      rowm[r] = fmaxf(v0, v1);
    }
#pragma unroll
    for (int off = 1; off < 16; off <<= 1)
#pragma unroll
      for (int r = 0; r < 8; ++r)
        rowm[r] = fmaxf(rowm[r], __shfl_xor(rowm[r], off, 32));

    float alpha[8], rowsum[8];
#pragma unroll
    for (int r = 0; r < 8; ++r) {
      float nm = fmaxf(m_r[r], rowm[r]);
      alpha[r] = __expf(m_r[r] - nm);
      m_r[r] = nm;
      float p0 = __expf(s0[r] - nm);
      float p1 = __expf(s1[r] - nm);
      s0[r] = p0; s1[r] = p1;
      rowsum[r] = p0 + p1;
    }
#pragma unroll
    for (int off = 1; off < 16; off <<= 1)
#pragma unroll
      for (int r = 0; r < 8; ++r)
        rowsum[r] += __shfl_xor(rowsum[r], off, 32);
#pragma unroll
    for (int r = 0; r < 8; ++r) l_r[r] = l_r[r] * alpha[r] + rowsum[r];

#pragma unroll
    for (int s = 0; s < 4; ++s)
#pragma unroll
      for (int r = 0; r < 8; ++r) o[s][r] *= alpha[r];

    // ---- P: C-layout fp32 -> A-fragment bf16 via wave-local LDS ----
#pragma unroll
    for (int r = 0; r < 8; ++r) {
      int row = r + 8 * (lane >> 4);
      pl[row * 32 + col]      = (__bf16)s0[r];
      pl[row * 32 + col + 16] = (__bf16)s1[r];
    }
    bf16x16 pa = load_a_frag(pl, 32);

    // ---- P @ V : B-fragment gathered from LDS V tile (keys x d) ----
#pragma unroll
    for (int s = 0; s < 4; ++s) {
      bf16x16 vb;
      int d    = s * 16 + col;
      int krel = (lane >> 4) << 4;
#pragma unroll
      for (int j = 0; j < 16; ++j)
        vb[j] = vtile[(size_t)(krel + j) * DHEAD + d];
      o[s] = wmma_bf16(pa, vb, o[s]);
    }
    __syncthreads();                    // tiles consumed before next TDM
  }

  // ---- epilogue: normalize, store context (chunked layout == flat BSD) ----
  __bf16* C = ctx + (size_t)bh * HEADELEM + (size_t)q0 * DHEAD;
  int half = lane >> 4;
#pragma unroll
  for (int s = 0; s < 4; ++s)
#pragma unroll
    for (int r = 0; r < 8; ++r) {
      float val = o[s][r] / l_r[r];
      C[(size_t)(r + 8 * half) * DHEAD + (s * 16 + col)] = (__bf16)val;
    }
}

// ---------------------------------------------------------------------------
extern "C" void kernel_launch(void* const* d_in, const int* in_sizes, int n_in,
                              void* d_out, int out_size, void* d_ws, size_t ws_size,
                              hipStream_t stream) {
  const float* q  = (const float*)d_in[0];
  const float* k  = (const float*)d_in[1];
  const float* v  = (const float*)d_in[2];
  const unsigned char* mask = (const unsigned char*)d_in[3];
  const float* wq = (const float*)d_in[4];
  const float* wk = (const float*)d_in[5];
  const float* wv = (const float*)d_in[6];
  const float* wo = (const float*)d_in[7];
  float* out = (float*)d_out;

  const size_t NQKV = (size_t)MROWS * DMODEL;
  const size_t NW   = (size_t)DMODEL * DMODEL;

  char* ws = (char*)d_ws;
  size_t off = 0;
  __bf16* qb  = (__bf16*)(ws + off); off += NQKV * 2;
  __bf16* kb  = (__bf16*)(ws + off); off += NQKV * 2;
  __bf16* vb  = (__bf16*)(ws + off); off += NQKV * 2;
  __bf16* wqb = (__bf16*)(ws + off); off += NW * 2;
  __bf16* wkb = (__bf16*)(ws + off); off += NW * 2;
  __bf16* wvb = (__bf16*)(ws + off); off += NW * 2;
  __bf16* wob = (__bf16*)(ws + off); off += NW * 2;
  __bf16* Qp  = (__bf16*)(ws + off); off += NQKV * 2;
  __bf16* Kp  = (__bf16*)(ws + off); off += NQKV * 2;
  __bf16* Vp  = (__bf16*)(ws + off); off += NQKV * 2;
  __bf16* ctx = (__bf16*)(ws + off); off += NQKV * 2;   // ~120 MiB total

  cvt_f32_to_bf16<<<(int)((NQKV + 255) / 256), 256, 0, stream>>>(q, qb, (int)NQKV);
  cvt_f32_to_bf16<<<(int)((NQKV + 255) / 256), 256, 0, stream>>>(k, kb, (int)NQKV);
  cvt_f32_to_bf16<<<(int)((NQKV + 255) / 256), 256, 0, stream>>>(v, vb, (int)NQKV);
  cvt_f32_to_bf16<<<(int)((NW   + 255) / 256), 256, 0, stream>>>(wq, wqb, (int)NW);
  cvt_f32_to_bf16<<<(int)((NW   + 255) / 256), 256, 0, stream>>>(wk, wkb, (int)NW);
  cvt_f32_to_bf16<<<(int)((NW   + 255) / 256), 256, 0, stream>>>(wv, wvb, (int)NW);
  cvt_f32_to_bf16<<<(int)((NW   + 255) / 256), 256, 0, stream>>>(wo, wob, (int)NW);

  dim3 ggrid(MROWS / 128, DMODEL / 64);
  gemm_xwt<<<ggrid, 256, 0, stream>>>(qb, wqb, nullptr, Qp, MROWS, DMODEL, DMODEL, 1);
  gemm_xwt<<<ggrid, 256, 0, stream>>>(kb, wkb, nullptr, Kp, MROWS, DMODEL, DMODEL, 1);
  gemm_xwt<<<ggrid, 256, 0, stream>>>(vb, wvb, nullptr, Vp, MROWS, DMODEL, DMODEL, 1);

  flash_attn<<<B_SZ * NHEAD * (S_LEN / 64), 128, 0, stream>>>(Qp, Kp, Vp, mask, ctx);

  gemm_xwt<<<ggrid, 256, 0, stream>>>(ctx, wob, out, nullptr, MROWS, DMODEL, DMODEL, 0);
}
